// PointTransformerSeg_40819369181271
// MI455X (gfx1250) — matrix operand order repaired and checked by reading, not compiled, not run on hardware
//
#include <hip/hip_runtime.h>
#include <hip/hip_bf16.h>

// ---------------------------------------------------------------------------
// Point Transformer segmentation forward pass for gfx1250 (MI455X).
// GEMMs: V_WMMA_F32_16X16X32_F16 (wave32), double-buffered pipeline:
//   A tiles DMA'd into LDS by the Tensor Data Mover (tensor_load_to_lds,
//   TENSORcnt) with hardware zero-fill for M/K tails and hardware row padding
//   (36-float stride) for bank-conflict-free ds_load_b128 fragment reads;
//   W tiles staged f32->f16 transposed in LDS. TDM(t+1) + W-stage(t+1)
//   overlap the WMMAs of tile t.
// ---------------------------------------------------------------------------

typedef long long i64;
typedef __attribute__((ext_vector_type(16))) _Float16 v16h;
typedef __attribute__((ext_vector_type(8)))  _Float16 v8h;
typedef __attribute__((ext_vector_type(8)))  float    v8f;
typedef unsigned int u32x4 __attribute__((ext_vector_type(4)));
typedef int          i32x4v __attribute__((ext_vector_type(4)));
typedef int          i32x8v __attribute__((ext_vector_type(8)));

static __device__ __forceinline__ float relu_f(float x) { return x > 0.f ? x : 0.f; }

union AFrag { v16h v; v8h h[2]; _Float16 e[16]; };
union CFrag { v8f  v; float e[8]; };

// ---------------------------------------------------------------------------
// TDM 2D tile load: global (row-major, f32, row stride `stride` elements)
// -> LDS at byte offset lds_off. tensor_dim0/1 = valid cols/rows remaining
// from the tile start (hardware zero-fills OOB). Row padding: pad_interval
// encodes DWORDs-before-pad (4 -> 32 DW), pad_amount DWORDs of pad (3 -> 4 DW)
// => effective LDS row stride 36 floats.
// ---------------------------------------------------------------------------
static __device__ __forceinline__ void tdm_load_2d(unsigned lds_off, const float* gptr,
                                                   int td0, int td1, int stride,
                                                   int tile0, int tile1,
                                                   int pad_int, int pad_amt)
{
    unsigned long long ga = (unsigned long long)(uintptr_t)gptr;
    u32x4 g0;
    g0[0] = 1u;                                           // count=1, user mode
    g0[1] = lds_off;                                      // lds_addr
    g0[2] = (unsigned)(ga & 0xFFFFFFFFull);               // global_addr[31:0]
    g0[3] = (unsigned)((ga >> 32) & 0x01FFFFFFull)        // global_addr[56:32]
          | (2u << 30);                                   // type=2 ("image")
    i32x8v g1;
    g1[0] = (2 << 16)                                     // data_size = 4B
          | (((pad_int | pad_amt) != 0) ? (1 << 20) : 0)  // pad_enable
          | (pad_int << 22) | (pad_amt << 25);
    g1[1] = (td0 & 0xFFFF) << 16;                         // tensor_dim0[15:0]
    g1[2] = ((td0 >> 16) & 0xFFFF) | ((td1 & 0xFFFF) << 16);
    g1[3] = ((td1 >> 16) & 0xFFFF) | ((tile0 & 0xFFFF) << 16);   // tile_dim0
    g1[4] = (tile1 & 0xFFFF);                             // tile_dim1 (tile_dim2=0)
    g1[5] = stride;                                       // tensor_dim0_stride[31:0]
    g1[6] = 0;
    g1[7] = 0;
    i32x4v z4 = {0, 0, 0, 0};
#if __clang_major__ >= 23
    i32x8v z8 = {0, 0, 0, 0, 0, 0, 0, 0};
    __builtin_amdgcn_tensor_load_to_lds(g0, g1, z4, z4, z8, 0);
#else
    __builtin_amdgcn_tensor_load_to_lds(g0, g1, z4, z4, 0);
#endif
}

// Stage one 32(K)x32(N) W tile as f16 transposed [n][k] (stride 40 halves).
// Thread t owns column nn=t&31, k rows {wave, wave+4, ..., wave+28}: walks
// rows by pointer increment (no per-element 64-bit muls), loads all 8 rows
// into temps first so the 8 global loads stay in flight together.
static __device__ __forceinline__ void stage_w_tile(const float* __restrict__ W,
                                                    int K, int N, int n0, int k0,
                                                    _Float16* lwp, int tid, int wave)
{
    const int nn = tid & 31;
    _Float16* dst = lwp + nn * 40 + wave;
    if (k0 + 32 <= K) {
        const float* src = W + (i64)(k0 + wave) * N + (n0 + nn);
        const i64 step = (i64)4 * N;
        float t[8];
#pragma unroll
        for (int r = 0; r < 8; ++r) { t[r] = *src; src += step; }
#pragma unroll
        for (int r = 0; r < 8; ++r) dst[4 * r] = (_Float16)t[r];
    } else {
        float t[8]; int ok[8];
#pragma unroll
        for (int r = 0; r < 8; ++r) {
            int kg = k0 + wave + 4 * r;
            int kc = kg < K ? kg : K - 1;                 // clamp: load stays in range
            ok[r] = kg < K;
            t[r] = W[(i64)kc * N + (n0 + nn)];
        }
#pragma unroll
        for (int r = 0; r < 8; ++r) dst[4 * r] = (_Float16)(ok[r] ? t[r] : 0.f);
    }
}

// ---------------------------------------------------------------------------
// Fast GEMM: C[M,N] = A[M,K] @ W[K,N] (+bias). Requires M%16==0, N%32==0.
// Block = 128 threads = 4 waves; block tile = 64 rows x 32 cols; k-step 32.
// Double-buffered: tile t+1 (TDM A + staged W) overlaps tile t's WMMAs.
// ---------------------------------------------------------------------------
__global__ void __launch_bounds__(128)
k_gemm_wmma_fast(const float* __restrict__ A, const float* __restrict__ W,
                 const float* __restrict__ bias, float* __restrict__ C,
                 i64 M, int K, int N)
{
    const int tid  = threadIdx.x;
    const int lane = tid & 31;
    const int wave = tid >> 5;
    const i64 mbase = (i64)blockIdx.y * 64;
    const i64 m0 = mbase + (i64)wave * 16;
    const int n0 = blockIdx.x * 32;
    const int half = lane >> 4;
    const int lm = lane & 15;
    const bool rowok = (m0 < M);          // wave-uniform

    __shared__ float     la[2][64 * 36];  // A tiles, 36-float row stride (TDM pad)
    __shared__ _Float16  lw[2][32 * 40];  // W tiles, [n][k] f16, stride 40 halves

    const unsigned lds_a[2] = { (unsigned)(uintptr_t)(&la[0][0]),
                                (unsigned)(uintptr_t)(&la[1][0]) };
    const float* Abase = A + mbase * (i64)K;
    const int mrem = (M - mbase) < 64 ? (int)(M - mbase) : 64;
    const int ntiles = (K + 31) / 32;

    CFrag acc0, acc1;
#pragma unroll
    for (int r = 0; r < 8; ++r) { acc0.e[r] = 0.f; acc1.e[r] = 0.f; }

    // ---- prologue: tile 0 ----
    if (wave == 0) {
        int krem = K < 32 ? K : 32;
        tdm_load_2d(lds_a[0], Abase, krem, mrem, K, 32, 64, 4, 3);
    }
    stage_w_tile(W, K, N, n0, 0, &lw[0][0], tid, wave);
    if (wave == 0) __builtin_amdgcn_s_wait_tensorcnt(0);
    __syncthreads();

    for (int t = 0; t < ntiles; ++t) {
        const int buf = t & 1;
        const int k0n = (t + 1) * 32;
        const bool more = (t + 1 < ntiles);

        // ---- issue tile t+1 into the other buffer ----
        if (more) {
            if (wave == 0) {
                int krem = (K - k0n) < 32 ? (K - k0n) : 32;
                tdm_load_2d(lds_a[buf ^ 1], Abase + k0n, krem, mrem, K, 32, 64, 4, 3);
            }
            stage_w_tile(W, K, N, n0, k0n, &lw[buf ^ 1][0], tid, wave);
        }

        // ---- compute tile t ----
        if (rowok) {
            AFrag a;
            {
                const float* ar0 = &la[buf][(wave * 16 + lm) * 36 + half * 8];
                float4 a0 = *(const float4*)(ar0);
                float4 a1 = *(const float4*)(ar0 + 4);
                float4 a2 = *(const float4*)(ar0 + 16);
                float4 a3 = *(const float4*)(ar0 + 20);
                a.e[0]=(_Float16)a0.x; a.e[1]=(_Float16)a0.y; a.e[2]=(_Float16)a0.z; a.e[3]=(_Float16)a0.w;
                a.e[4]=(_Float16)a1.x; a.e[5]=(_Float16)a1.y; a.e[6]=(_Float16)a1.z; a.e[7]=(_Float16)a1.w;
                a.e[8]=(_Float16)a2.x; a.e[9]=(_Float16)a2.y; a.e[10]=(_Float16)a2.z; a.e[11]=(_Float16)a2.w;
                a.e[12]=(_Float16)a3.x; a.e[13]=(_Float16)a3.y; a.e[14]=(_Float16)a3.z; a.e[15]=(_Float16)a3.w;
            }
            AFrag b0, b1;
            {
                const _Float16* s0 = &lw[buf][lm * 40 + half * 8];
                const _Float16* s1 = &lw[buf][(lm + 16) * 40 + half * 8];
                b0.h[0] = *(const v8h*)(s0);
                b0.h[1] = *(const v8h*)(s0 + 16);
                b1.h[0] = *(const v8h*)(s1);
                b1.h[1] = *(const v8h*)(s1 + 16);
            }
            acc0.v = __builtin_amdgcn_wmma_f32_16x16x32_f16(
                false, a.v, false, b0.v, (short)0, acc0.v, false, false);
            acc1.v = __builtin_amdgcn_wmma_f32_16x16x32_f16(
                false, a.v, false, b1.v, (short)0, acc1.v, false, false);
        }

        if (more && wave == 0) __builtin_amdgcn_s_wait_tensorcnt(0);
        __syncthreads();
    }

    if (rowok) {
        const int nc0 = n0 + lm, nc1 = n0 + 16 + lm;
        const float bv0 = bias ? bias[nc0] : 0.f;
        const float bv1 = bias ? bias[nc1] : 0.f;
#pragma unroll
        for (int r = 0; r < 8; ++r) {
            const i64 mr = m0 + r + half * 8;
            C[mr * N + nc0] = acc0.e[r] + bv0;
            C[mr * N + nc1] = acc1.e[r] + bv1;
        }
    }
}

// ---------------------------------------------------------------------------
// Generic GEMM (edge shapes: N%32!=0 or M%16!=0). Clamped-address loads +
// selects instead of predicated loads (no exec-mask dance).
// ---------------------------------------------------------------------------
__global__ void __launch_bounds__(128)
k_gemm_wmma_gen(const float* __restrict__ A, const float* __restrict__ W,
                const float* __restrict__ bias, float* __restrict__ C,
                i64 M, int K, int N)
{
    const int lane = threadIdx.x & 31;
    const int wave = threadIdx.x >> 5;
    const i64 m0 = (i64)blockIdx.y * 64 + (i64)wave * 16;
    const int n0 = blockIdx.x * 16;
    if (m0 >= M) return;                       // wave-uniform

    const int  half = lane >> 4;
    const i64  mrow = m0 + (lane & 15);
    const i64  mc   = mrow < M ? mrow : M - 1;
    const bool mok  = (mrow < M);
    const int  ncol = n0 + (lane & 15);
    const int  ncc  = ncol < N ? ncol : N - 1;
    const bool nok  = (ncol < N);

    CFrag acc;
#pragma unroll
    for (int r = 0; r < 8; ++r) acc.e[r] = 0.f;

    for (int k0 = 0; k0 < K; k0 += 32) {
        AFrag a, b;
#pragma unroll
        for (int i = 0; i < 16; ++i) {
            const int ka = k0 + ((i >= 8) ? 16 : 0) + half * 8 + (i & 7);
            const int kac = ka < K ? ka : K - 1;
            float av = A[mc * (i64)K + kac];
            a.e[i] = (_Float16)((mok && ka < K) ? av : 0.f);
            const int kb = k0 + ((i >= 8) ? 16 : 0) + half * 8 + (i & 7);
            const int kbc = kb < K ? kb : K - 1;
            float wv = W[(i64)kbc * N + ncc];
            b.e[i] = (_Float16)((nok && kb < K) ? wv : 0.f);
        }
        acc.v = __builtin_amdgcn_wmma_f32_16x16x32_f16(
            false, a.v, false, b.v, (short)0, acc.v, false, false);
    }

    const float bv = (bias != nullptr && nok) ? bias[ncol] : 0.f;
#pragma unroll
    for (int r = 0; r < 8; ++r) {
        const i64 mr = m0 + r + half * 8;
        if (mr < M && nok) C[mr * N + ncol] = acc.e[r] + bv;
    }
}

// ---------------------------------------------------------------------------
// BatchNorm (batch stats): pass 1 per-channel mean/var, pass 2 normalize.
// ---------------------------------------------------------------------------
__global__ void k_bn_stats(const float* __restrict__ X, i64 rows, int C,
                           float* __restrict__ mean, float* __restrict__ var)
{
    const int c = blockIdx.x, tid = threadIdx.x;
    float s = 0.f, s2 = 0.f;
    for (i64 r = tid; r < rows; r += blockDim.x) {
        float v = X[r * C + c];
        s += v; s2 += v * v;
    }
    __shared__ float ss[256], sq[256];
    ss[tid] = s; sq[tid] = s2; __syncthreads();
    for (int o = 128; o > 0; o >>= 1) {
        if (tid < o) { ss[tid] += ss[tid + o]; sq[tid] += sq[tid + o]; }
        __syncthreads();
    }
    if (tid == 0) {
        float m = ss[0] / (float)rows;
        mean[c] = m;
        var[c]  = sq[0] / (float)rows - m * m;
    }
}

__global__ void k_bn_apply(float* __restrict__ X, i64 total, int C,
                           const float* __restrict__ mean, const float* __restrict__ var,
                           const float* __restrict__ g, const float* __restrict__ b,
                           int do_relu)
{
    i64 t = (i64)blockIdx.x * blockDim.x + threadIdx.x;
    if (t >= total) return;
    int c = (int)(t % C);
    float inv = rsqrtf(var[c] + 1e-5f);
    float y = (X[t] - mean[c]) * inv * g[c] + b[c];
    X[t] = do_relu ? relu_f(y) : y;
}

// ---------------------------------------------------------------------------
// KNN: per query thread, register-resident sorted top-KT over LDS-tiled points.
// ---------------------------------------------------------------------------
template <int KT>
__global__ void __launch_bounds__(128)
k_knn(const float* __restrict__ Q, int Nq, const float* __restrict__ P, int Np,
      int skip_first, int* __restrict__ idx_out, float* __restrict__ d2_out)
{
    const int b  = blockIdx.y;
    const int qi = blockIdx.x * blockDim.x + threadIdx.x;
    const bool act = (qi < Nq);
    float qx = 0.f, qy = 0.f, qz = 0.f;
    if (act) {
        const float* qp = Q + ((i64)b * Nq + qi) * 3;
        qx = qp[0]; qy = qp[1]; qz = qp[2];
    }
    float bd[KT]; int bi[KT];
#pragma unroll
    for (int j = 0; j < KT; ++j) { bd[j] = 3.0e38f; bi[j] = 0; }

    __shared__ float sp[128 * 3];
    for (int t0 = 0; t0 < Np; t0 += 128) {
        int tn = Np - t0; if (tn > 128) tn = 128;
        __syncthreads();
        if ((int)threadIdx.x < tn) {
            const float* pp = P + ((i64)b * Np + t0 + threadIdx.x) * 3;
            sp[threadIdx.x * 3 + 0] = pp[0];
            sp[threadIdx.x * 3 + 1] = pp[1];
            sp[threadIdx.x * 3 + 2] = pp[2];
        }
        __syncthreads();
        if (act) {
            for (int j = 0; j < tn; ++j) {
                float dx = qx - sp[j * 3 + 0];
                float dy = qy - sp[j * 3 + 1];
                float dz = qz - sp[j * 3 + 2];
                float d = dx * dx + dy * dy + dz * dz;
                if (d < bd[KT - 1]) {
                    bd[KT - 1] = d; bi[KT - 1] = t0 + j;
#pragma unroll
                    for (int s = KT - 1; s > 0; --s) {
                        if (bd[s] < bd[s - 1]) {
                            float td = bd[s]; bd[s] = bd[s - 1]; bd[s - 1] = td;
                            int   ti = bi[s]; bi[s] = bi[s - 1]; bi[s - 1] = ti;
                        }
                    }
                }
            }
        }
    }
    if (act) {
        const int ko = KT - skip_first;
#pragma unroll
        for (int j = 0; j < KT; ++j) {
            if (j >= skip_first) {
                i64 o = ((i64)b * Nq + qi) * ko + (j - skip_first);
                idx_out[o] = bi[j];
                if (d2_out) d2_out[o] = bd[j];
            }
        }
    }
}

// ---------------------------------------------------------------------------
// Farthest point sampling: one workgroup per batch, LDS argmax reduction.
// ---------------------------------------------------------------------------
__global__ void __launch_bounds__(256)
k_fps(const float* __restrict__ P, int Np, int M,
      int* __restrict__ sidx, float* __restrict__ newp, float* __restrict__ mind_ws)
{
    const int b = blockIdx.x, tid = threadIdx.x;
    float* mind = mind_ws + (i64)b * Np;
    for (int i = tid; i < Np; i += 256) mind[i] = 1e10f;
    __shared__ float sv[256];
    __shared__ int   si[256];
    __shared__ int   s_last;
    if (tid == 0) s_last = 0;
    __syncthreads();
    for (int it = 0; it < M; ++it) {
        const int last = s_last;
        const float* lp = P + ((i64)b * Np + last) * 3;
        const float lx = lp[0], ly = lp[1], lz = lp[2];
        if (tid == 0) {
            sidx[(i64)b * M + it] = last;
            float* np_ = newp + ((i64)b * M + it) * 3;
            np_[0] = lx; np_[1] = ly; np_[2] = lz;
        }
        float bm = -1.f; int bj = 0;
        for (int i = tid; i < Np; i += 256) {
            const float* pp = P + ((i64)b * Np + i) * 3;
            float dx = pp[0] - lx, dy = pp[1] - ly, dz = pp[2] - lz;
            float d = dx * dx + dy * dy + dz * dz;
            float m = mind[i]; m = (d < m) ? d : m; mind[i] = m;
            if (m > bm) { bm = m; bj = i; }
        }
        sv[tid] = bm; si[tid] = bj; __syncthreads();
        for (int o = 128; o > 0; o >>= 1) {
            if (tid < o && sv[tid + o] > sv[tid]) { sv[tid] = sv[tid + o]; si[tid] = si[tid + o]; }
            __syncthreads();
        }
        if (tid == 0) s_last = si[0];
        __syncthreads();
    }
}

// ---------------------------------------------------------------------------
// Gather / elementwise helpers
// ---------------------------------------------------------------------------
__global__ void k_gather_feat(const float* __restrict__ x, const int* __restrict__ idx,
                              float* __restrict__ out, int Nq, int K, int C, int Np, i64 total)
{
    i64 t = (i64)blockIdx.x * blockDim.x + threadIdx.x;
    if (t >= total) return;
    int c = (int)(t % C);
    i64 r = t / C;                       // (b, q, k) row
    i64 b = r / ((i64)Nq * K);
    int id = idx[r];
    out[t] = x[((i64)b * Np + id) * C + c];
}

__global__ void k_gather_sub3(const float* __restrict__ Pdb, const int* __restrict__ idx,
                              const float* __restrict__ Pq, float* __restrict__ out,
                              int Nq, int K, int Np, i64 total)
{
    i64 t = (i64)blockIdx.x * blockDim.x + threadIdx.x;
    if (t >= total) return;
    i64 b = t / ((i64)Nq * K);
    i64 q = (t / K) % Nq;
    int id = idx[t];
    const float* pd = Pdb + ((i64)b * Np + id) * 3;
    const float* pq = Pq + ((i64)b * Nq + q) * 3;
    float* o = out + t * 3;
    o[0] = pd[0] - pq[0]; o[1] = pd[1] - pq[1]; o[2] = pd[2] - pq[2];
}

__global__ void k_gather_rows_i(const int* __restrict__ src, const int* __restrict__ sidx,
                                int* __restrict__ dst, int M, int Np, int K, i64 total)
{
    i64 t = (i64)blockIdx.x * blockDim.x + threadIdx.x;
    if (t >= total) return;
    int k = (int)(t % K);
    i64 m = (t / K) % M;
    i64 b = t / ((i64)M * K);
    dst[t] = src[((i64)b * Np + sidx[b * M + m]) * K + k];
}

__global__ void k_w_combine(const float* __restrict__ q, const float* __restrict__ xk,
                            const float* __restrict__ pr, float* __restrict__ w,
                            int C, i64 total)
{
    i64 t = (i64)blockIdx.x * blockDim.x + threadIdx.x;
    if (t >= total) return;
    int c = (int)(t % C);
    i64 row = t / C;            // (b*n*16 + k)
    i64 bn = row / 16;
    w[t] = q[bn * C + c] - xk[t] + pr[t];
}

__global__ void k_softmax_wsum(const float* __restrict__ w, const float* __restrict__ xv,
                               const float* __restrict__ pr, float* __restrict__ out,
                               i64 M, int C)
{
    i64 t = (i64)blockIdx.x * blockDim.x + threadIdx.x;
    if (t >= M * C) return;
    i64 row = t / C;
    int c = (int)(t % C);
    const float* wp = w + (row * 16) * (i64)C + c;
    float mx = -3.0e38f;
#pragma unroll
    for (int k = 0; k < 16; ++k) { float v = wp[(i64)k * C]; mx = v > mx ? v : mx; }
    float sum = 0.f, acc = 0.f;
#pragma unroll
    for (int k = 0; k < 16; ++k) {
        float e = __expf(wp[(i64)k * C] - mx);
        sum += e;
        i64 o = (row * 16 + k) * (i64)C + c;
        acc += (xv[o] + pr[o]) * e;
    }
    out[t] = acc / sum;
}

__global__ void k_max_over_k(const float* __restrict__ x, float* __restrict__ y,
                             i64 M, int C)
{
    i64 t = (i64)blockIdx.x * blockDim.x + threadIdx.x;
    if (t >= M * C) return;
    i64 row = t / C;
    int c = (int)(t % C);
    float m = -3.0e38f;
#pragma unroll
    for (int k = 0; k < 16; ++k) {
        float v = x[(row * 16 + k) * (i64)C + c];
        m = v > m ? v : m;
    }
    y[t] = m;
}

__global__ void k_copy_cols(const float* __restrict__ src, int srcC,
                            float* __restrict__ dst, int dstC, int off, i64 total)
{
    i64 t = (i64)blockIdx.x * blockDim.x + threadIdx.x;
    if (t >= total) return;
    i64 r = t / srcC;
    int c = (int)(t % srcC);
    dst[r * dstC + off + c] = src[t];
}

__global__ void k_copy_bcast(const float* __restrict__ g, int C,
                             float* __restrict__ dst, int dstC, int off,
                             int Nrows, i64 total)
{
    i64 t = (i64)blockIdx.x * blockDim.x + threadIdx.x;
    if (t >= total) return;
    int c = (int)(t % C);
    i64 r = (t / C) % Nrows;
    i64 b = t / ((i64)Nrows * C);
    dst[((i64)b * Nrows + r) * dstC + off + c] = g[b * C + c];
}

__global__ void k_mean_rows(const float* __restrict__ x, float* __restrict__ g,
                            int Nrows, int C)
{
    const int b = blockIdx.x / C, c = blockIdx.x % C, tid = threadIdx.x;
    float s = 0.f;
    for (int r = tid; r < Nrows; r += blockDim.x) s += x[((i64)b * Nrows + r) * C + c];
    __shared__ float ss[256];
    ss[tid] = s; __syncthreads();
    for (int o = 128; o > 0; o >>= 1) {
        if (tid < o) ss[tid] += ss[tid + o];
        __syncthreads();
    }
    if (tid == 0) g[(i64)b * C + c] = ss[0] / (float)Nrows;
}

__global__ void k_interp_apply(const float* __restrict__ feat, const int* __restrict__ idx,
                               const float* __restrict__ d2, float* __restrict__ out,
                               int Nd, int Nc, int C, i64 total)
{
    i64 t = (i64)blockIdx.x * blockDim.x + threadIdx.x;
    if (t >= total) return;
    int c = (int)(t % C);
    i64 n = (t / C) % Nd;
    i64 b = t / ((i64)Nd * C);
    i64 base = ((i64)b * Nd + n) * 3;
    float w0 = 1.f / (d2[base + 0] + 1e-8f);
    float w1 = 1.f / (d2[base + 1] + 1e-8f);
    float w2 = 1.f / (d2[base + 2] + 1e-8f);
    float inv = 1.f / (w0 + w1 + w2);
    float acc = w0 * feat[((i64)b * Nc + idx[base + 0]) * C + c]
              + w1 * feat[((i64)b * Nc + idx[base + 1]) * C + c]
              + w2 * feat[((i64)b * Nc + idx[base + 2]) * C + c];
    out[t] = acc * inv;
}

__global__ void k_relu(float* __restrict__ x, i64 n)
{
    i64 t = (i64)blockIdx.x * blockDim.x + threadIdx.x;
    if (t < n) x[t] = relu_f(x[t]);
}

__global__ void k_add(const float* __restrict__ a, const float* __restrict__ b,
                      float* __restrict__ y, i64 n)
{
    i64 t = (i64)blockIdx.x * blockDim.x + threadIdx.x;
    if (t < n) y[t] = a[t] + b[t];
}

__global__ void k_add_relu(const float* __restrict__ a, const float* __restrict__ b,
                           float* __restrict__ y, i64 n)
{
    i64 t = (i64)blockIdx.x * blockDim.x + threadIdx.x;
    if (t < n) y[t] = relu_f(a[t] + b[t]);
}

// ===========================================================================
// Host-side orchestration
// ===========================================================================
static constexpr int Bb = 2;

static inline unsigned cdiv(i64 a, i64 b) { return (unsigned)((a + b - 1) / b); }

struct Arena {
    char*  base;
    size_t off, cap;
    float* f(size_t n)  { size_t b = (off + 255) & ~(size_t)255; off = b + n * sizeof(float); return (float*)(base + b); }
    int*   i32(size_t n){ size_t b = (off + 255) & ~(size_t)255; off = b + n * sizeof(int);   return (int*)(base + b); }
};

struct LinP { const float* w; const float* b; };
struct BNP  { const float* g; const float* b; };
struct PtlP { LinP q, k, v, p1, p2, w1, w2; BNP p_bn, w_bn1, w_bn2; };
struct BlkP { LinP l1, l3; BNP bn1, bn2, bn3; PtlP tr; };
struct TDP  { LinP lin; BNP bn; };
struct UpP  { LinP l1, l2; BNP bn1, bn2; };

// jax pytree flatten order: dict keys sorted alphabetically, recursively.
struct Cur {
    void* const* in; int i;
    const float* nx() { return (const float*)in[i++]; }
};
static LinP ld_lin_b(Cur& c) { LinP l; l.b = c.nx(); l.w = c.nx(); return l; }   // {'b','w'}
static LinP ld_lin(Cur& c)   { LinP l; l.b = nullptr; l.w = c.nx(); return l; } // {'w'}
static BNP  ld_bn(Cur& c)    { BNP b; b.b = c.nx(); b.g = c.nx(); return b; }   // {'b','g'}
static PtlP ld_ptl(Cur& c) { // k, p1, p2, p_bn, q, v, w1, w2, w_bn1, w_bn2
    PtlP p;
    p.k = ld_lin_b(c); p.p1 = ld_lin_b(c); p.p2 = ld_lin_b(c); p.p_bn = ld_bn(c);
    p.q = ld_lin_b(c); p.v = ld_lin_b(c);
    p.w1 = ld_lin_b(c); p.w2 = ld_lin_b(c); p.w_bn1 = ld_bn(c); p.w_bn2 = ld_bn(c);
    return p;
}
static BlkP ld_blk(Cur& c) { // bn1, bn2, bn3, l1, l3, tr
    BlkP b;
    b.bn1 = ld_bn(c); b.bn2 = ld_bn(c); b.bn3 = ld_bn(c);
    b.l1 = ld_lin(c); b.l3 = ld_lin(c); b.tr = ld_ptl(c);
    return b;
}
static UpP ld_up(Cur& c) {  // bn1, bn2, l1, l2
    UpP u; u.bn1 = ld_bn(c); u.bn2 = ld_bn(c); u.l1 = ld_lin_b(c); u.l2 = ld_lin_b(c); return u;
}
static UpP ld_up5(Cur& c) { // bn1, l1, l2
    UpP u; u.bn1 = ld_bn(c); u.bn2.g = nullptr; u.bn2.b = nullptr;
    u.l1 = ld_lin_b(c); u.l2 = ld_lin_b(c); return u;
}
static TDP ld_td(Cur& c) {  // bn, lin
    TDP t; t.bn = ld_bn(c); t.lin = ld_lin(c); return t;
}

// ------------------- building-block launchers -------------------
static void linear(hipStream_t st, const float* X, i64 M, int K, int N,
                   const LinP& L, float* Y)
{
    if ((N & 31) == 0 && (M & 15) == 0) {
        dim3 grid(cdiv(N, 32), cdiv(M, 64));
        k_gemm_wmma_fast<<<grid, 128, 0, st>>>(X, L.w, L.b, Y, M, K, N);
    } else {
        dim3 grid(cdiv(N, 16), cdiv(M, 64));
        k_gemm_wmma_gen<<<grid, 128, 0, st>>>(X, L.w, L.b, Y, M, K, N);
    }
}

static void bn_relu(hipStream_t st, float* X, i64 rows, int C, const BNP& bnp,
                    bool do_relu, Arena s)
{
    float* mv = s.f(2 * (size_t)C);
    k_bn_stats<<<C, 256, 0, st>>>(X, rows, C, mv, mv + C);
    i64 tot = rows * C;
    k_bn_apply<<<cdiv(tot, 256), 256, 0, st>>>(X, tot, C, mv, mv + C, bnp.g, bnp.b,
                                               do_relu ? 1 : 0);
}

static void ptlayer(hipStream_t st, const float* x, const float* p3, int Npts, int c,
                    const PtlP& P, float* out, int* idx_out, Arena s)
{
    const i64 M = (i64)Bb * Npts, MK = M * 16;
    float* qb = s.f(M * c); float* kb = s.f(M * c); float* vb = s.f(M * c);
    linear(st, x, M, c, c, P.q, qb);
    linear(st, x, M, c, c, P.k, kb);
    linear(st, x, M, c, c, P.v, vb);
    k_knn<17><<<dim3(cdiv(Npts, 128), Bb), 128, 0, st>>>(p3, Npts, p3, Npts, 1,
                                                         idx_out, nullptr);
    float* xk = s.f(MK * c);
    k_gather_feat<<<cdiv(MK * c, 256), 256, 0, st>>>(kb, idx_out, xk, Npts, 16, c, Npts, MK * c);
    float* xv = s.f(MK * c);
    k_gather_feat<<<cdiv(MK * c, 256), 256, 0, st>>>(vb, idx_out, xv, Npts, 16, c, Npts, MK * c);
    float* prin = s.f(MK * 3);
    k_gather_sub3<<<cdiv(MK, 256), 256, 0, st>>>(p3, idx_out, p3, prin, Npts, 16, Npts, MK);
    float* pr3 = s.f(MK * 3);
    linear(st, prin, MK, 3, 3, P.p1, pr3);
    bn_relu(st, pr3, MK, 3, P.p_bn, true, s);
    float* prc = s.f(MK * c);
    linear(st, pr3, MK, 3, c, P.p2, prc);
    float* w = s.f(MK * c);
    k_w_combine<<<cdiv(MK * c, 256), 256, 0, st>>>(qb, xk, prc, w, c, MK * c);
    bn_relu(st, w, MK, c, P.w_bn1, true, s);
    float* wt = s.f(MK * c);
    linear(st, w, MK, c, c, P.w1, wt);
    bn_relu(st, wt, MK, c, P.w_bn2, true, s);
    linear(st, wt, MK, c, c, P.w2, w);
    k_softmax_wsum<<<cdiv(M * c, 256), 256, 0, st>>>(w, xv, prc, out, M, c);
}

static void block(hipStream_t st, const float* x, const float* p3, int Npts, int c,
                  const BlkP& Bk, float* out, int* ind_out, Arena s)
{
    const i64 M = (i64)Bb * Npts;
    float* h = s.f(M * c);
    linear(st, x, M, c, c, Bk.l1, h);
    bn_relu(st, h, M, c, Bk.bn1, true, s);
    float* h2 = s.f(M * c);
    ptlayer(st, h, p3, Npts, c, Bk.tr, h2, ind_out, s);
    bn_relu(st, h2, M, c, Bk.bn2, true, s);
    float* h3 = s.f(M * c);
    linear(st, h2, M, c, c, Bk.l3, h3);
    bn_relu(st, h3, M, c, Bk.bn3, false, s);
    k_add_relu<<<cdiv(M * c, 256), 256, 0, st>>>(h3, x, out, M * c);
}

static void enc_down(hipStream_t st, const float* x, const float* pp, const int* knn_ind,
                     int Np, int cin, int cout, const TDP& td, const BlkP& blk,
                     float* xo, float* new_p, int* ind_out, Arena s)
{
    const int M = Np / 4;
    int* sidx = s.i32((size_t)Bb * M);
    float* mind = s.f((size_t)Bb * Np);
    k_fps<<<Bb, 256, 0, st>>>(pp, Np, M, sidx, new_p, mind);
    int* nn = s.i32((size_t)Bb * M * 16);
    k_gather_rows_i<<<cdiv((i64)Bb * M * 16, 256), 256, 0, st>>>(knn_ind, sidx, nn, M, Np, 16,
                                                                 (i64)Bb * M * 16);
    const i64 R = (i64)Bb * M * 16;
    float* feat = s.f(R * cin);
    k_gather_feat<<<cdiv(R * cin, 256), 256, 0, st>>>(x, nn, feat, M, 16, cin, Np, R * cin);
    float* gxyz = s.f(R * 3);
    k_gather_sub3<<<cdiv(R, 256), 256, 0, st>>>(pp, nn, new_p, gxyz, M, 16, Np, R);
    const int cf = 3 + cin;
    float* f = s.f(R * cf);
    k_copy_cols<<<cdiv(R * 3, 256), 256, 0, st>>>(gxyz, 3, f, cf, 0, R * 3);
    k_copy_cols<<<cdiv(R * cin, 256), 256, 0, st>>>(feat, cin, f, cf, 3, R * cin);
    float* h = s.f(R * cout);
    linear(st, f, R, cf, cout, td.lin, h);
    bn_relu(st, h, R, cout, td.bn, true, s);
    float* hm = s.f((size_t)Bb * M * cout);
    k_max_over_k<<<cdiv((i64)Bb * M * cout, 256), 256, 0, st>>>(h, hm, (i64)Bb * M, cout);
    block(st, hm, new_p, M, cout, blk, xo, ind_out, s);
}

static void interp(hipStream_t st, const float* pc, const float* pd, const float* feat,
                   int Nd, int Nc, int C, float* out, Arena s)
{
    int* idx = s.i32((size_t)Bb * Nd * 3);
    float* d2 = s.f((size_t)Bb * Nd * 3);
    k_knn<3><<<dim3(cdiv(Nd, 128), Bb), 128, 0, st>>>(pd, Nd, pc, Nc, 0, idx, d2);
    i64 tot = (i64)Bb * Nd * C;
    k_interp_apply<<<cdiv(tot, 256), 256, 0, st>>>(feat, idx, d2, out, Nd, Nc, C, tot);
}

static void up_path(hipStream_t st, const float* x1, const float* p1, int Nd,
                    const float* x2, const float* p2, int Nc, int c,
                    const UpP& U, float* out, Arena s)
{
    const i64 Md = (i64)Bb * Nd, Mc = (i64)Bb * Nc;
    float* h1 = s.f(Md * c);
    linear(st, x1, Md, c, c, U.l1, h1);
    bn_relu(st, h1, Md, c, U.bn1, true, s);
    float* h2 = s.f(Mc * c);
    linear(st, x2, Mc, 2 * c, c, U.l2, h2);   // x2 has 2c channels
    bn_relu(st, h2, Mc, c, U.bn2, true, s);
    float* it = s.f(Md * c);
    interp(st, p2, p1, h2, Nd, Nc, c, it, s);
    k_add<<<cdiv(Md * c, 256), 256, 0, st>>>(h1, it, out, Md * c);
}

static void up_head(hipStream_t st, const float* x, int Npts, int c, const UpP& U,
                    float* out, Arena s)
{
    float* g = s.f((size_t)Bb * c);
    k_mean_rows<<<Bb * c, 256, 0, st>>>(x, g, Npts, c);
    float* g2 = s.f((size_t)Bb * c);
    linear(st, g, Bb, c, c, U.l2, g2);
    k_relu<<<cdiv((i64)Bb * c, 256), 256, 0, st>>>(g2, (i64)Bb * c);
    const i64 M = (i64)Bb * Npts;
    float* hc = s.f((size_t)M * 2 * c);
    k_copy_cols<<<cdiv(M * c, 256), 256, 0, st>>>(x, c, hc, 2 * c, 0, M * c);
    k_copy_bcast<<<cdiv(M * c, 256), 256, 0, st>>>(g2, c, hc, 2 * c, c, Npts, M * c);
    linear(st, hc, M, 2 * c, c, U.l1, out);
    bn_relu(st, out, M, c, U.bn1, true, s);
}

// ------------------- top level -------------------
extern "C" void kernel_launch(void* const* d_in, const int* in_sizes, int n_in,
                              void* d_out, int out_size, void* d_ws, size_t ws_size,
                              hipStream_t stream)
{
    (void)in_sizes; (void)n_in; (void)out_size;
    const float* P = (const float*)d_in[0];   // (B, 8192, 3)

    // Unpack params (alphabetical pytree order).
    Cur c{d_in, 1};
    BlkP dec1_blk = ld_blk(c); UpP dec1_up = ld_up(c);
    BlkP dec2_blk = ld_blk(c); UpP dec2_up = ld_up(c);
    BlkP dec3_blk = ld_blk(c); UpP dec3_up = ld_up(c);
    BlkP dec4_blk = ld_blk(c); UpP dec4_up = ld_up(c);
    BlkP dec5_blk = ld_blk(c); UpP dec5_up = ld_up5(c);
    BlkP enc1_blk = ld_blk(c); TDP enc1_td = ld_td(c);
    BlkP enc2_blk = ld_blk(c); TDP enc2_td = ld_td(c);
    BlkP enc3_blk = ld_blk(c); TDP enc3_td = ld_td(c);
    BlkP enc4_blk = ld_blk(c); TDP enc4_td = ld_td(c);
    BlkP enc5_blk = ld_blk(c); TDP enc5_td = ld_td(c);
    LinP seg1 = ld_lin_b(c); LinP seg2 = ld_lin_b(c); BNP seg_bn = ld_bn(c);

    Arena ar{(char*)d_ws, 0, ws_size};
    const int N1 = 8192, N2 = 2048, N3 = 512, N4 = 128, N5 = 32;
    const i64 M1 = (i64)Bb * N1;

    // ---- encoder ----
    float* h0 = ar.f((size_t)M1 * 32);
    { Arena s = ar; linear(stream, P, M1, 3, 32, enc1_td.lin, h0);
      bn_relu(stream, h0, M1, 32, enc1_td.bn, true, s); }
    float* x1 = ar.f((size_t)M1 * 32);
    int* i1 = ar.i32((size_t)M1 * 16);
    { Arena s = ar; block(stream, h0, P, N1, 32, enc1_blk, x1, i1, s); }

    float* p2 = ar.f((size_t)Bb * N2 * 3); float* x2 = ar.f((size_t)Bb * N2 * 64);
    int* i2 = ar.i32((size_t)Bb * N2 * 16);
    { Arena s = ar; enc_down(stream, x1, P, i1, N1, 32, 64, enc2_td, enc2_blk, x2, p2, i2, s); }

    float* p3 = ar.f((size_t)Bb * N3 * 3); float* x3 = ar.f((size_t)Bb * N3 * 128);
    int* i3 = ar.i32((size_t)Bb * N3 * 16);
    { Arena s = ar; enc_down(stream, x2, p2, i2, N2, 64, 128, enc3_td, enc3_blk, x3, p3, i3, s); }

    float* p4 = ar.f((size_t)Bb * N4 * 3); float* x4 = ar.f((size_t)Bb * N4 * 256);
    int* i4 = ar.i32((size_t)Bb * N4 * 16);
    { Arena s = ar; enc_down(stream, x3, p3, i3, N3, 128, 256, enc4_td, enc4_blk, x4, p4, i4, s); }

    float* p5 = ar.f((size_t)Bb * N5 * 3); float* x5 = ar.f((size_t)Bb * N5 * 512);
    int* i5 = ar.i32((size_t)Bb * N5 * 16);
    { Arena s = ar; enc_down(stream, x4, p4, i4, N4, 256, 512, enc5_td, enc5_blk, x5, p5, i5, s); }

    // ---- decoder ----
    float* d5in = ar.f((size_t)Bb * N5 * 512);
    { Arena s = ar; up_head(stream, x5, N5, 512, dec5_up, d5in, s); }
    float* x5d = ar.f((size_t)Bb * N5 * 512);
    { Arena s = ar; int* it = s.i32((size_t)Bb * N5 * 16);
      block(stream, d5in, p5, N5, 512, dec5_blk, x5d, it, s); }

    float* u4 = ar.f((size_t)Bb * N4 * 256);
    { Arena s = ar; up_path(stream, x4, p4, N4, x5d, p5, N5, 256, dec4_up, u4, s); }
    float* x4d = ar.f((size_t)Bb * N4 * 256);
    { Arena s = ar; int* it = s.i32((size_t)Bb * N4 * 16);
      block(stream, u4, p4, N4, 256, dec4_blk, x4d, it, s); }

    float* u3 = ar.f((size_t)Bb * N3 * 128);
    { Arena s = ar; up_path(stream, x3, p3, N3, x4d, p4, N4, 128, dec3_up, u3, s); }
    float* x3d = ar.f((size_t)Bb * N3 * 128);
    { Arena s = ar; int* it = s.i32((size_t)Bb * N3 * 16);
      block(stream, u3, p3, N3, 128, dec3_blk, x3d, it, s); }

    float* u2 = ar.f((size_t)Bb * N2 * 64);
    { Arena s = ar; up_path(stream, x2, p2, N2, x3d, p3, N3, 64, dec2_up, u2, s); }
    float* x2d = ar.f((size_t)Bb * N2 * 64);
    { Arena s = ar; int* it = s.i32((size_t)Bb * N2 * 16);
      block(stream, u2, p2, N2, 64, dec2_blk, x2d, it, s); }

    float* u1 = ar.f((size_t)M1 * 32);
    { Arena s = ar; up_path(stream, x1, P, N1, x2d, p2, N2, 32, dec1_up, u1, s); }
    float* x1d = ar.f((size_t)M1 * 32);
    { Arena s = ar; int* it = s.i32((size_t)M1 * 16);
      block(stream, u1, P, N1, 32, dec1_blk, x1d, it, s); }

    // ---- segmentation head ----
    float* hseg = ar.f((size_t)M1 * 32);
    { Arena s = ar; linear(stream, x1d, M1, 32, 32, seg1, hseg);
      bn_relu(stream, hseg, M1, 32, seg_bn, true, s); }
    linear(stream, hseg, M1, 32, 6, seg2, (float*)d_out);
}
